// GNNModel_22265110462801
// MI455X (gfx1250) — matrix operand order repaired and checked by reading, not compiled, not run on hardware
//
#include <hip/hip_runtime.h>
#include <hip/hip_bf16.h>
#include <cstdint>

// Problem constants (match reference)
#define NN   10000
#define EE   40000
#define CEIN 16
#define CEH  128

typedef __attribute__((ext_vector_type(16))) _Float16 v16h;
typedef __attribute__((ext_vector_type(8)))  _Float16 v8h;
typedef __attribute__((ext_vector_type(8)))  float    v8f;

// ---------------------------------------------------------------------------
// Edge MLP stage 1: h[e][j] = relu(eattr[e,:16] @ w1[:,j] + b1[j]) -> f16
// ---------------------------------------------------------------------------
__global__ __launch_bounds__(256)
void edge_hidden_kernel(const float* __restrict__ eattr,  // [E,16]
                        const float* __restrict__ w1,     // [16,128]
                        const float* __restrict__ b1,     // [128]
                        _Float16* __restrict__ h)         // [E,128] f16
{
    int idx = blockIdx.x * blockDim.x + threadIdx.x;
    if (idx >= EE * CEH) return;
    int e = idx >> 7;
    int j = idx & 127;
    const float* er = eattr + (size_t)e * CEIN;
    float acc = b1[j];
#pragma unroll
    for (int k = 0; k < CEIN; ++k)
        acc += er[k] * w1[k * CEH + j];
    acc = acc > 0.f ? acc : 0.f;
    h[idx] = (_Float16)acc;
}

// ---------------------------------------------------------------------------
// Transpose + convert: w2 [128, 2048] f32 -> w2t [2048, 128] f16
// ---------------------------------------------------------------------------
__global__ __launch_bounds__(256)
void w2_transpose_kernel(const float* __restrict__ w2,
                         _Float16* __restrict__ w2t)
{
    int idx = blockIdx.x * blockDim.x + threadIdx.x;   // n*128 + k
    if (idx >= 2048 * 128) return;
    int n = idx >> 7;
    int k = idx & 127;
    w2t[idx] = (_Float16)w2[(size_t)k * 2048 + n];
}

// ---------------------------------------------------------------------------
// Fused NNConv message kernel.
// Each wave owns MT=2 sub-tiles of 16 edges (32 edges total): every B tile
// (32x16 slice of relu(h@w2+b2)'s weight producer) is loaded once and feeds
// two v_wmma_f32_16x16x32_f16 issues, halving L2 B-traffic per FLOP.
// The WMMA accumulator fragment is consumed in-register: + b2, ReLU, scale by
// x[src[e], i] (from LDS), accumulated into per-wave msg fragments, then
// scattered with global_atomic_add_f32 into aggr[dst].
// E = 40000 = 1250 tiles of 32 edges = 625 blocks x 2 waves (no tail).
// ---------------------------------------------------------------------------
template<int CIN, int COUT>
__global__ __launch_bounds__(64)
void nnconv_edge_wmma_kernel(const _Float16* __restrict__ h,    // [E,128] f16
                             const _Float16* __restrict__ w2t,  // [CIN*COUT,128] f16
                             const float*    __restrict__ b2,   // [CIN*COUT]
                             const float*    __restrict__ xin,  // [N,CIN]
                             const int*      __restrict__ src,  // [E]
                             const int*      __restrict__ dst,  // [E]
                             float*          __restrict__ aggr) // [N,COUT]
{
    constexpr int WAVES  = 2;
    constexpr int MT     = 2;           // 16-edge sub-tiles per wave
    constexpr int GROUPS = COUT / 16;

    __shared__ float xs[WAVES][MT * 16 * CIN];

    const int lane = threadIdx.x & 31;
    const int wid  = threadIdx.x >> 5;
    const int tile = blockIdx.x * WAVES + wid;   // 0..1249, exact
    const int e0   = tile * (MT * 16);

    // Stage x[src] rows for this wave's 32 edges into LDS (same-wave DS ops
    // are in-order; this wave is the only reader of its region).
    for (int t = lane; t < MT * 16 * CIN; t += 32) {
        int er = t / CIN, c = t - er * CIN;
        xs[wid][t] = xin[(size_t)src[e0 + er] * CIN + c];
    }

    const int m  = lane & 15;     // A row / C column selector
    const int hi = lane >> 4;     // half-wave selector

    // A fragments: h rows, MT sub-tiles x 4 K-tiles.
    // 16-bit A 16x32 layout: lanes 0-15 -> M=lane, K = {0..7, 16..23};
    //                        lanes 16-31 -> M=lane-16, K = {8..15, 24..31}.
    v16h a[MT][4];
#pragma unroll
    for (int s = 0; s < MT; ++s) {
        const _Float16* hrow = h + (size_t)(e0 + s * 16 + m) * 128;
#pragma unroll
        for (int k = 0; k < 4; ++k) {
            int k0 = k * 32 + hi * 8;
            v8h lo = *(const v8h*)(hrow + k0);
            v8h hh = *(const v8h*)(hrow + k0 + 16);
#pragma unroll
            for (int t = 0; t < 8; ++t) { a[s][k][t] = lo[t]; a[s][k][8 + t] = hh[t]; }
        }
    }

    // msg accumulators (C-fragment layout) per sub-tile per 16-column group
    v8f msg[MT][GROUPS];
#pragma unroll
    for (int s = 0; s < MT; ++s)
#pragma unroll
        for (int g = 0; g < GROUPS; ++g)
#pragma unroll
            for (int t = 0; t < 8; ++t) msg[s][g][t] = 0.f;

    for (int i = 0; i < CIN; ++i) {
        // x[src[e], i] for the 8 C-rows this lane holds, per sub-tile
        float xv[MT][8];
#pragma unroll
        for (int s = 0; s < MT; ++s)
#pragma unroll
            for (int v = 0; v < 8; ++v)
                xv[s][v] = xs[wid][(s * 16 + v + hi * 8) * CIN + i];

#pragma unroll
        for (int g = 0; g < GROUPS; ++g) {
            const int n0 = i * COUT + g * 16;   // column base in [0,2048)
            v8f c[MT];
#pragma unroll
            for (int s = 0; s < MT; ++s)
#pragma unroll
                for (int t = 0; t < 8; ++t) c[s][t] = 0.f;

#pragma unroll
            for (int k = 0; k < 4; ++k) {
                // B (32x16) 16-bit layout: lanes 0-15 -> col=lane, K = {0..7, 16..23};
                //                          lanes 16-31 -> col=lane-16, K = {8..15, 24..31}.
                const _Float16* bcol =
                    w2t + (size_t)(n0 + m) * 128 + k * 32 + hi * 8;
                v8h b0 = *(const v8h*)(bcol);
                v8h b1 = *(const v8h*)(bcol + 16);
                v16h b;
#pragma unroll
                for (int t = 0; t < 8; ++t) { b[t] = b0[t]; b[8 + t] = b1[t]; }
                // One B tile feeds MT WMMAs (B reuse across edge sub-tiles)
#pragma unroll
                for (int s = 0; s < MT; ++s)
                    c[s] = __builtin_amdgcn_wmma_f32_16x16x32_f16(
                               false, a[s][k], false, b, (short)0, c[s],
                               false, false);
            }

            const float bias = b2[n0 + m];
#pragma unroll
            for (int s = 0; s < MT; ++s)
#pragma unroll
                for (int v = 0; v < 8; ++v) {
                    float w = c[s][v] + bias;
                    w = w > 0.f ? w : 0.f;           // relu on edge weights
                    msg[s][g][v] += xv[s][v] * w;     // einsum contraction
                }
        }
    }

    // segment_sum: scatter msg into aggr[dst]
#pragma unroll
    for (int s = 0; s < MT; ++s)
#pragma unroll
        for (int v = 0; v < 8; ++v) {
            const int d = dst[e0 + s * 16 + v + hi * 8];
#pragma unroll
            for (int g = 0; g < GROUPS; ++g)
                atomicAdd(&aggr[(size_t)d * COUT + g * 16 + m], msg[s][g][v]);
        }
}

// ---------------------------------------------------------------------------
// Node update: out[n,j] = act(aggr[n,j] + x[n,:] @ root[:,j] + bias[j])
// ---------------------------------------------------------------------------
template<int CIN, int COUT, bool RELU>
__global__ __launch_bounds__(256)
void node_update_kernel(const float* __restrict__ aggr,  // [N,COUT]
                        const float* __restrict__ xin,   // [N,CIN]
                        const float* __restrict__ root,  // [CIN,COUT]
                        const float* __restrict__ bias,  // [COUT]
                        float* __restrict__ out)         // [N,COUT]
{
    int idx = blockIdx.x * blockDim.x + threadIdx.x;
    if (idx >= NN * COUT) return;
    int n = idx / COUT;
    int j = idx - n * COUT;
    float acc = aggr[idx] + bias[j];
    const float* xr = xin + (size_t)n * CIN;
#pragma unroll 8
    for (int k = 0; k < CIN; ++k)
        acc += xr[k] * root[k * COUT + j];
    if (RELU) acc = acc > 0.f ? acc : 0.f;
    out[idx] = acc;
}

// ---------------------------------------------------------------------------
// Launch
// ---------------------------------------------------------------------------
extern "C" void kernel_launch(void* const* d_in, const int* in_sizes, int n_in,
                              void* d_out, int out_size, void* d_ws, size_t ws_size,
                              hipStream_t stream)
{
    (void)in_sizes; (void)n_in; (void)out_size; (void)ws_size;

    const float* x       = (const float*)d_in[0];
    const int*   eidx    = (const int*)  d_in[1];
    const float* eattr   = (const float*)d_in[2];
    const float* l1_ew1  = (const float*)d_in[3];
    const float* l1_eb1  = (const float*)d_in[4];
    const float* l1_ew2  = (const float*)d_in[5];
    const float* l1_eb2  = (const float*)d_in[6];
    const float* l1_root = (const float*)d_in[7];
    const float* l1_bias = (const float*)d_in[8];
    const float* l2_ew1  = (const float*)d_in[9];
    const float* l2_eb1  = (const float*)d_in[10];
    const float* l2_ew2  = (const float*)d_in[11];
    const float* l2_eb2  = (const float*)d_in[12];
    const float* l2_root = (const float*)d_in[13];
    const float* l2_bias = (const float*)d_in[14];

    const int* src = eidx;         // edge_index[0]
    const int* dst = eidx + EE;    // edge_index[1]

    // Workspace layout (all sizes 256B-aligned)
    char* ws = (char*)d_ws;
    _Float16* h16  = (_Float16*)ws;  ws += (size_t)EE * CEH * sizeof(_Float16); // 10.24 MB
    _Float16* w2t1 = (_Float16*)ws;  ws += (size_t)2048 * 128 * sizeof(_Float16);
    _Float16* w2t2 = (_Float16*)ws;  ws += (size_t)2048 * 128 * sizeof(_Float16);
    float* aggr1   = (float*)ws;     ws += (size_t)NN * 64 * sizeof(float);
    float* x1      = (float*)ws;     ws += (size_t)NN * 64 * sizeof(float);
    float* aggr2   = (float*)ws;     ws += (size_t)NN * 32 * sizeof(float);

    hipMemsetAsync(aggr1, 0, (size_t)NN * 64 * sizeof(float), stream);
    hipMemsetAsync(aggr2, 0, (size_t)NN * 32 * sizeof(float), stream);

    // Weight transposes (f32 -> f16, [128,2048] -> [2048,128])
    w2_transpose_kernel<<<(2048 * 128 + 255) / 256, 256, 0, stream>>>(l1_ew2, w2t1);
    w2_transpose_kernel<<<(2048 * 128 + 255) / 256, 256, 0, stream>>>(l2_ew2, w2t2);

    const int nblk_tile = (EE / 32) / 2;   // 1250 tiles / 2 waves = 625

    // ---- Layer 1: 32 -> 64 ----
    edge_hidden_kernel<<<(EE * CEH + 255) / 256, 256, 0, stream>>>(
        eattr, l1_ew1, l1_eb1, h16);
    nnconv_edge_wmma_kernel<32, 64><<<nblk_tile, 64, 0, stream>>>(
        h16, w2t1, l1_eb2, x, src, dst, aggr1);
    node_update_kernel<32, 64, true><<<(NN * 64 + 255) / 256, 256, 0, stream>>>(
        aggr1, x, l1_root, l1_bias, x1);

    // ---- Layer 2: 64 -> 32 ----
    edge_hidden_kernel<<<(EE * CEH + 255) / 256, 256, 0, stream>>>(
        eattr, l2_ew1, l2_eb1, h16);
    nnconv_edge_wmma_kernel<64, 32><<<nblk_tile, 64, 0, stream>>>(
        h16, w2t2, l2_eb2, x1, src, dst, aggr2);
    node_update_kernel<64, 32, false><<<(NN * 32 + 255) / 256, 256, 0, stream>>>(
        aggr2, x1, l2_root, l2_bias, (float*)d_out);
}
